// Multiheaded_Attention_22548578304253
// MI455X (gfx1250) — matrix operand order repaired and checked by reading, not compiled, not run on hardware
//
#include <hip/hip_runtime.h>
#include <hip/hip_bf16.h>

// ---------------------------------------------------------------------------
// Multi-head attention for MI455X (gfx1250, wave32, WMMA f32_16x16x32_f16)
// ---------------------------------------------------------------------------
typedef _Float16 half_t;
typedef __attribute__((ext_vector_type(16))) _Float16 v16h;
typedef __attribute__((ext_vector_type(8)))  _Float16 v8h;
typedef __attribute__((ext_vector_type(8)))  float    v8f;

#define NB    4
#define SEQ   1024
#define DM    1024
#define NHEAD 16
#define HD    64

// ---- WMMA fragment loaders (contiguous-k order, same for A and B, wave32) ------------
// A (16x32 f16, MxK): lane L holds row m=L&15; with g=L>>4 the 16 halfs are
//   K = g*16 .. g*16+15 (contiguous)
__device__ __forceinline__ v16h load_a_frag(const half_t* p, int ld, int lane) {
  const int m = lane & 15, g = lane >> 4;
  const half_t* row = p + m * ld;
  union { v16h v; v8h h[2]; } u;
  u.h[0] = *(const v8h*)(row + g * 16);
  u.h[1] = *(const v8h*)(row + g * 16 + 8);
  return u.v;
}
// B (32x16 f16, KxN) staged as BT[n][k]: lane L holds column n=L&15,
//   K = (L>>4)*16 .. +15  -> 16 contiguous halfs (32B)
__device__ __forceinline__ v16h load_b_frag(const half_t* p, int ld, int lane) {
  const int n = lane & 15, g = lane >> 4;
  return *(const v16h*)(p + n * ld + g * 16);
}
__device__ __forceinline__ v8f wmma_f16(v16h a, v16h b, v8f c) {
  return __builtin_amdgcn_wmma_f32_16x16x32_f16(false, a, false, b,
                                                (short)0, c, false, false);
}

// ---------------------------------------------------------------------------
// Kernel 1: projection  out = (A[4096x1024] @ W[1024x1024] + bias) * scale
// Output f16, head-split layout [b][h][s][64].
// Block = 256 threads (8 waves), tile 128x128, K-chunk 32.
// ---------------------------------------------------------------------------
__global__ __launch_bounds__(256)
void proj_kernel(const float* __restrict__ A, const float* __restrict__ W,
                 const float* __restrict__ bias, half_t* __restrict__ outh,
                 float scale) {
  __shared__ __align__(32) half_t As[128 * 32];
  __shared__ __align__(32) half_t Bt[128 * 32];
  const int t = threadIdx.x, lane = t & 31, w = t >> 5;
  const int mbase = blockIdx.x * 128;
  const int nbase = blockIdx.y * 128;

  v8f zero = {};
  v8f acc[8];
#pragma unroll
  for (int i = 0; i < 8; ++i) acc[i] = zero;

  for (int kb = 0; kb < DM; kb += 32) {
    __syncthreads();
    // A tile: 128x32 f32 -> f16 (coalesced float4)
#pragma unroll
    for (int j = 0; j < 4; ++j) {
      int f4 = t + 256 * j;                 // 0..1023
      int row = f4 >> 3, c4 = f4 & 7;
      const float4 va = *(const float4*)(A + (size_t)(mbase + row) * DM + kb + c4 * 4);
      half_t* d = As + row * 32 + c4 * 4;
      d[0] = (half_t)va.x; d[1] = (half_t)va.y;
      d[2] = (half_t)va.z; d[3] = (half_t)va.w;
    }
    // B tile: 32x128 f32, transposed into Bt[n][k]
#pragma unroll
    for (int j = 0; j < 4; ++j) {
      int f4 = t + 256 * j;
      int k = f4 >> 5, n4 = f4 & 31;
      const float4 vb = *(const float4*)(W + (size_t)(kb + k) * DM + nbase + n4 * 4);
      Bt[(n4 * 4 + 0) * 32 + k] = (half_t)vb.x;
      Bt[(n4 * 4 + 1) * 32 + k] = (half_t)vb.y;
      Bt[(n4 * 4 + 2) * 32 + k] = (half_t)vb.z;
      Bt[(n4 * 4 + 3) * 32 + k] = (half_t)vb.w;
    }
    __syncthreads();
    v16h a = load_a_frag(As + w * 16 * 32, 32, lane);
#pragma unroll
    for (int nt = 0; nt < 8; ++nt) {
      v16h b = load_b_frag(Bt + nt * 16 * 32, 32, lane);
      acc[nt] = wmma_f16(a, b, acc[nt]);
    }
  }
  // epilogue: head-split f16
  const int g = lane >> 4, nlo = lane & 15;
#pragma unroll
  for (int nt = 0; nt < 8; ++nt) {
    const int col = nbase + nt * 16 + nlo;
    const int h = col >> 6, d = col & 63;
    const float bv = bias[col];
#pragma unroll
    for (int r = 0; r < 8; ++r) {
      const int row = mbase + w * 16 + r + 8 * g;
      const int b = row >> 10, s = row & 1023;
      const float val = (acc[nt][r] + bv) * scale;
      outh[((size_t)(b * NHEAD + h) * SEQ + s) * HD + d] = (half_t)val;
    }
  }
}

// ---------------------------------------------------------------------------
// Kernel 2: scores  S[bh][q][kv] = q[bh,q,:] . k[bh,kv,:]   (K = 64)
// Writes fp32 pre-softmax scores directly into attention_scores output.
// ---------------------------------------------------------------------------
__global__ __launch_bounds__(256)
void scores_kernel(const half_t* __restrict__ qh, const half_t* __restrict__ kh,
                   float* __restrict__ scoresOut) {
  __shared__ __align__(32) half_t Qs[128 * 64];
  __shared__ __align__(32) half_t Ks[128 * 64];
  const int t = threadIdx.x, lane = t & 31, w = t >> 5;
  const int bh = blockIdx.x;
  const int qbase = blockIdx.y * 128;
  const int nbase = blockIdx.z * 128;

  // both tiles are fully contiguous 16KB regions: flat uint4 copy
  const uint4* qsrc = (const uint4*)(qh + ((size_t)bh * SEQ + qbase) * HD);
  const uint4* ksrc = (const uint4*)(kh + ((size_t)bh * SEQ + nbase) * HD);
  uint4* qdst = (uint4*)Qs;
  uint4* kdst = (uint4*)Ks;
#pragma unroll
  for (int j = 0; j < 4; ++j) {
    qdst[t + 256 * j] = qsrc[t + 256 * j];
    kdst[t + 256 * j] = ksrc[t + 256 * j];
  }
  __syncthreads();

  v8f zero = {};
  v8f acc[8];
#pragma unroll
  for (int i = 0; i < 8; ++i) acc[i] = zero;
#pragma unroll
  for (int kc = 0; kc < 64; kc += 32) {
    v16h a = load_a_frag(Qs + w * 16 * 64 + kc, 64, lane);
#pragma unroll
    for (int nt = 0; nt < 8; ++nt) {
      // Ks rows ([kv][64]) are already the BT[n][k] layout
      v16h b = load_b_frag(Ks + nt * 16 * 64 + kc, 64, lane);
      acc[nt] = wmma_f16(a, b, acc[nt]);
    }
  }
  const int g = lane >> 4, nlo = lane & 15;
#pragma unroll
  for (int nt = 0; nt < 8; ++nt)
#pragma unroll
    for (int r = 0; r < 8; ++r) {
      const int qrow = qbase + w * 16 + r + 8 * g;
      const int col = nbase + nt * 16 + nlo;
      scoresOut[((size_t)bh * SEQ + qrow) * SEQ + col] = acc[nt][r];
    }
}

// ---------------------------------------------------------------------------
// Kernel 3: masked softmax over the key dim, in place. One block per row.
// ---------------------------------------------------------------------------
__global__ __launch_bounds__(256)
void softmax_kernel(float* __restrict__ scores, const float* __restrict__ mask) {
  __shared__ float red[256];
  const int t = threadIdx.x;
  const int rid = blockIdx.x;          // bh*SEQ + q
  const int bh = rid >> 10;
  const int q = rid & 1023;
  const int b = bh >> 4;
  float* row = scores + (size_t)rid * SEQ;
  const float* mrow = mask + ((size_t)b * SEQ + q) * SEQ;

  float vals[4];
  float mx = -3.4e38f;
#pragma unroll
  for (int j = 0; j < 4; ++j) {
    const int c = t + 256 * j;
    const float s = row[c];
    vals[j] = (mrow[c] != 0.0f) ? s : -__builtin_inff();
    mx = fmaxf(mx, vals[j]);
  }
  red[t] = mx;
  __syncthreads();
  for (int s = 128; s > 0; s >>= 1) {
    if (t < s) red[t] = fmaxf(red[t], red[t + s]);
    __syncthreads();
  }
  mx = red[0];
  __syncthreads();

  float sum = 0.0f;
#pragma unroll
  for (int j = 0; j < 4; ++j) {
    vals[j] = __expf(vals[j] - mx);
    sum += vals[j];
  }
  red[t] = sum;
  __syncthreads();
  for (int s = 128; s > 0; s >>= 1) {
    if (t < s) red[t] += red[t + s];
    __syncthreads();
  }
  const float inv = 1.0f / red[0];
#pragma unroll
  for (int j = 0; j < 4; ++j) row[t + 256 * j] = vals[j] * inv;
}

// ---------------------------------------------------------------------------
// Kernel 4: context  O[bh] = P[bh] (1024x1024 fp32) @ V[bh] (1024x64 f16)
// Output f16 context in [b][s][h*64+n] layout for the final GEMM.
// ---------------------------------------------------------------------------
__global__ __launch_bounds__(256)
void pv_kernel(const float* __restrict__ attn, const half_t* __restrict__ vh,
               half_t* __restrict__ ctx) {
  __shared__ __align__(32) half_t Ps[128 * 32];
  __shared__ __align__(32) half_t Vt[64 * 32];
  const int t = threadIdx.x, lane = t & 31, w = t >> 5;
  const int bh = blockIdx.x;
  const int mbase = blockIdx.y * 128;
  const int bb = bh >> 4, h = bh & 15;

  v8f zero = {};
  v8f acc[4];
#pragma unroll
  for (int i = 0; i < 4; ++i) acc[i] = zero;

  for (int kb = 0; kb < SEQ; kb += 32) {
    __syncthreads();
    // P tile 128x32 fp32 -> f16
#pragma unroll
    for (int j = 0; j < 4; ++j) {
      int f4 = t + 256 * j;
      int row = f4 >> 3, c4 = f4 & 7;
      const float4 va =
          *(const float4*)(attn + ((size_t)bh * SEQ + mbase + row) * SEQ + kb + c4 * 4);
      half_t* d = Ps + row * 32 + c4 * 4;
      d[0] = (half_t)va.x; d[1] = (half_t)va.y;
      d[2] = (half_t)va.z; d[3] = (half_t)va.w;
    }
    // V tile 32x64 f16 -> transposed Vt[n][k]
    {
      const int k = t >> 3, c8 = t & 7;
      const v8h hv = *(const v8h*)(vh + ((size_t)bh * SEQ + kb + k) * HD + c8 * 8);
#pragma unroll
      for (int u = 0; u < 8; ++u) Vt[(c8 * 8 + u) * 32 + k] = hv[u];
    }
    __syncthreads();
    v16h a = load_a_frag(Ps + w * 16 * 32, 32, lane);
#pragma unroll
    for (int nt = 0; nt < 4; ++nt) {
      v16h b = load_b_frag(Vt + nt * 16 * 32, 32, lane);
      acc[nt] = wmma_f16(a, b, acc[nt]);
    }
  }
  const int g = lane >> 4, nlo = lane & 15;
#pragma unroll
  for (int nt = 0; nt < 4; ++nt)
#pragma unroll
    for (int r = 0; r < 8; ++r) {
      const int s = mbase + w * 16 + r + 8 * g;
      const int col = nt * 16 + nlo;
      ctx[((size_t)bb * SEQ + s) * DM + h * HD + col] = (half_t)acc[nt][r];
    }
}

// ---------------------------------------------------------------------------
// Kernel 5: attended = ctx(f16, 4096x1024) @ Wc + bc  -> fp32 output
// ---------------------------------------------------------------------------
__global__ __launch_bounds__(256)
void final_kernel(const half_t* __restrict__ ctx, const float* __restrict__ Wc,
                  const float* __restrict__ bc, float* __restrict__ out) {
  __shared__ __align__(32) half_t As[128 * 32];
  __shared__ __align__(32) half_t Bt[128 * 32];
  const int t = threadIdx.x, lane = t & 31, w = t >> 5;
  const int mbase = blockIdx.x * 128;
  const int nbase = blockIdx.y * 128;

  v8f zero = {};
  v8f acc[8];
#pragma unroll
  for (int i = 0; i < 8; ++i) acc[i] = zero;

  for (int kb = 0; kb < DM; kb += 32) {
    __syncthreads();
    // A tile: 128x32 halves (512 x 16B)
#pragma unroll
    for (int j = 0; j < 2; ++j) {
      int c8 = t + 256 * j;                 // 0..511
      int row = c8 >> 2, c = c8 & 3;
      *(uint4*)(As + row * 32 + c * 8) =
          *(const uint4*)(ctx + (size_t)(mbase + row) * DM + kb + c * 8);
    }
    // B tile: 32x128 f32 -> transposed Bt[n][k]
#pragma unroll
    for (int j = 0; j < 4; ++j) {
      int f4 = t + 256 * j;
      int k = f4 >> 5, n4 = f4 & 31;
      const float4 vb = *(const float4*)(Wc + (size_t)(kb + k) * DM + nbase + n4 * 4);
      Bt[(n4 * 4 + 0) * 32 + k] = (half_t)vb.x;
      Bt[(n4 * 4 + 1) * 32 + k] = (half_t)vb.y;
      Bt[(n4 * 4 + 2) * 32 + k] = (half_t)vb.z;
      Bt[(n4 * 4 + 3) * 32 + k] = (half_t)vb.w;
    }
    __syncthreads();
    v16h a = load_a_frag(As + w * 16 * 32, 32, lane);
#pragma unroll
    for (int nt = 0; nt < 8; ++nt) {
      v16h b = load_b_frag(Bt + nt * 16 * 32, 32, lane);
      acc[nt] = wmma_f16(a, b, acc[nt]);
    }
  }
  const int g = lane >> 4, nlo = lane & 15;
#pragma unroll
  for (int nt = 0; nt < 8; ++nt) {
    const int col = nbase + nt * 16 + nlo;
    const float bv = bc[col];
#pragma unroll
    for (int r = 0; r < 8; ++r) {
      const int row = mbase + w * 16 + r + 8 * g;
      out[(size_t)row * DM + col] = acc[nt][r] + bv;
    }
  }
}

// ---------------------------------------------------------------------------
extern "C" void kernel_launch(void* const* d_in, const int* in_sizes, int n_in,
                              void* d_out, int out_size, void* d_ws, size_t ws_size,
                              hipStream_t stream) {
  (void)in_sizes; (void)n_in; (void)out_size; (void)ws_size;

  const float* Q    = (const float*)d_in[0];
  const float* K    = (const float*)d_in[1];
  const float* V    = (const float*)d_in[2];
  const float* mask = (const float*)d_in[3];
  const float* Wq   = (const float*)d_in[4];
  const float* bq   = (const float*)d_in[5];
  const float* Wk   = (const float*)d_in[6];
  const float* bk   = (const float*)d_in[7];
  const float* Wv   = (const float*)d_in[8];
  const float* bv   = (const float*)d_in[9];
  const float* Wc   = (const float*)d_in[10];
  const float* bc   = (const float*)d_in[11];

  float* attended = (float*)d_out;                               // [4,1024,1024]
  float* scores   = attended + (size_t)NB * SEQ * DM;            // [4,16,1024,1024]

  const size_t headElems = (size_t)NB * NHEAD * SEQ * HD;        // 4M halves
  half_t* qh  = (half_t*)d_ws;
  half_t* kh  = qh + headElems;
  half_t* vh  = kh + headElems;
  half_t* ctx = vh + headElems;                                  // [4,1024,1024] f16

  const dim3 blk(256);
  const dim3 gproj(32, 8);   // 4096/128 x 1024/128
  const float scaling = 0.125f;  // HD^-0.5

  proj_kernel<<<gproj, blk, 0, stream>>>(Q, Wq, bq, qh, scaling);
  proj_kernel<<<gproj, blk, 0, stream>>>(K, Wk, bk, kh, 1.0f);
  proj_kernel<<<gproj, blk, 0, stream>>>(V, Wv, bv, vh, 1.0f);
  scores_kernel<<<dim3(NB * NHEAD, 8, 8), blk, 0, stream>>>(qh, kh, scores);
  softmax_kernel<<<dim3(NB * NHEAD * SEQ), blk, 0, stream>>>(scores, mask);
  pv_kernel<<<dim3(NB * NHEAD, 8), blk, 0, stream>>>(scores, vh, ctx);
  final_kernel<<<gproj, blk, 0, stream>>>(ctx, Wc, bc, attended);
}
